// Model_76579266888241
// MI455X (gfx1250) — matrix operand order repaired and checked by reading, compiled, not run on hardware
//
#include <hip/hip_runtime.h>

// ---------------- problem constants ----------------
#define BB      64
#define LL      512
#define EE      512            // E == H == H2
#define NHEAD   8
#define NWG     28             // persistent workgroups
#define NTHR    256            // 8 wave32 per WG
#define KTILES  16             // 512 / 32
#define NT_A    128            // [Wq|Wh] = 2048 cols / 16
#define NT_TOT  224            // + Wx 1536/16 = 96

// ---------------- workspace layout (bytes) ----------------
#define OFF_CNT   0u
#define OFF_HF16  1024u                       // h   f16 [64][512]   64KB
#define OFF_HF32  66560u                      // h   f32 [64][512]  128KB
#define OFF_HSUM  197632u                     // sum f32 [64][512]  128KB
#define OFF_U     328704u                     // U   f32 [64][2048] 512KB
#define OFF_GX    852992u                     // gx  f32 [64][1536] 384KB
#define OFF_CTX   1246208u                    // ctx f16 [64][512]   64KB
#define OFF_T1    1311744u                    // fc1 f32 [64][512]  128KB
#define OFF_PW    1442816u                    // packed weights f16 3.5MB

typedef __attribute__((ext_vector_type(16))) _Float16 v16h;
typedef __attribute__((ext_vector_type(8)))  float    v8f;

union Frag { uint4 u[2]; v16h h; };           // 32B: one WMMA A/B operand

__device__ __forceinline__ unsigned short f2h(float f) {
  _Float16 h = (_Float16)f;
  union { _Float16 h; unsigned short s; } u; u.h = h; return u.s;
}
__device__ __forceinline__ float sigm(float v) { return 1.0f / (1.0f + __expf(-v)); }

// grid-wide barrier: monotonic counter, reset by memset each launch
__device__ __forceinline__ void gridBarrier(unsigned* cnt, unsigned target) {
  __syncthreads();
  if (threadIdx.x == 0) {
    __threadfence();                       // release my stores
    atomicAdd(cnt, 1u);
    while (atomicAdd(cnt, 0u) < target) __builtin_amdgcn_s_sleep(1);
    __threadfence();                       // acquire before fresh reads
  }
  __syncthreads();
}

// ---------------- weight packing: f32 -> f16 in WMMA B-fragment order ----------------
// frag dword index = ((T*16 + kb)*32 + lane)*8 + v
// element (K = kb*32 + (lane>>4)*16 + 2v + p,  N = T*16 + (lane&15)), p packed lo/hi
__global__ void pack_weights(const float* __restrict__ Wq,
                             const float* __restrict__ Wh,
                             const float* __restrict__ Wx,
                             unsigned* __restrict__ dst) {
  int idx = blockIdx.x * NTHR + threadIdx.x;
  if (idx >= NT_TOT * 4096) return;
  int v    = idx & 7;
  int lane = (idx >> 3) & 31;
  int kb   = (idx >> 8) & 15;
  int T    = idx >> 12;
  int k    = kb * 32 + ((lane >> 4) << 4) + 2 * v;
  int nl   = lane & 15;
  float a, b;
  if (T < NT_A) {                 // concat [Wq | Wh] along N (2048 cols)
    int c = T * 16 + nl;
    if (c < 512) { a = Wq[k * 512 + c];          b = Wq[(k + 1) * 512 + c]; }
    else         { a = Wh[k * 1536 + (c - 512)]; b = Wh[(k + 1) * 1536 + (c - 512)]; }
  } else {                        // Wx (1536 cols)
    int c = (T - NT_A) * 16 + nl;
    a = Wx[k * 1536 + c];  b = Wx[(k + 1) * 1536 + c];
  }
  dst[idx] = (unsigned)f2h(a) | ((unsigned)f2h(b) << 16);
}

// ---------------- persistent scan kernel ----------------
__global__ __launch_bounds__(NTHR)
void scan_kernel(const int*   __restrict__ x,     // [64][512]
                 const float* __restrict__ emb,   // [50000][512]
                 const float* __restrict__ bx, const float* __restrict__ bh,
                 const float* __restrict__ W1, const float* __restrict__ b1,
                 const float* __restrict__ W2, const float* __restrict__ b2,
                 const unsigned* __restrict__ packedW,
                 unsigned* cnt,
                 _Float16* hF16, float* hF32, float* hsum,
                 float* U, float* gx, _Float16* ctx,
                 float* t1, float* out) {
  extern __shared__ char smem[];
  uint4* ldsW = (uint4*)smem;                     // 8 tiles * 1024 uint4 = 128KB
  float* qlds = (float*)(smem + 131072);          // [64][128] f32 = 32KB (WGs 0..3)

  const int wg   = blockIdx.x;                    // 0..27
  const int tid  = threadIdx.x;
  const int wave = tid >> 5;                      // wave32
  const int lane = tid & 31;
  const int T    = wg * 8 + wave;                 // global n-tile 0..223
  const bool isA = (T < NT_A);

  // weights -> LDS once, reused for all 512 steps
  {
    const uint4* src = (const uint4*)packedW + (size_t)wg * 8192;
    for (int i = tid; i < 8192; i += NTHR) ldsW[i] = src[i];
  }
  __syncthreads();

  unsigned ep = 0;
  const int gtid = wg * NTHR + tid;               // 0..7167

  for (int t = 0; t < LL; ++t) {
    // ---------- Phase A: U = h @ [Wq|Wh]  (WGs 0..15) ----------
    if (isA) {
      v8f acc[4];
      for (int m = 0; m < 4; ++m) for (int i = 0; i < 8; ++i) acc[m][i] = 0.0f;
      for (int kb = 0; kb < KTILES; ++kb) {
        Frag bf;
        const uint4* bp = &ldsW[(size_t)wave * 1024 + (kb * 32 + lane) * 2];
        bf.u[0] = bp[0]; bf.u[1] = bp[1];
        for (int m = 0; m < 4; ++m) {
          Frag af;
          const char* abase = (const char*)hF16 +
              (size_t)(m * 16 + (lane & 15)) * 1024 + kb * 64 + ((lane >> 4) << 5);
          af.u[0] = *(const uint4*)abase;
          af.u[1] = *(const uint4*)(abase + 16);
          acc[m] = __builtin_amdgcn_wmma_f32_16x16x32_f16(
              false, af.h, false, bf.h, (short)0, acc[m], false, false);
        }
      }
      for (int m = 0; m < 4; ++m) {
        int row = m * 16 + ((lane >> 4) << 3);
        int col = T * 16 + (lane & 15);
        for (int r = 0; r < 8; ++r) {
          float val = acc[m][r];
          U[(size_t)(row + r) * 2048 + col] = val;
          if (wg < 4) qlds[(row + r) * 128 + (col - wg * 128)] = val;   // q slice
        }
      }
    }
    __syncthreads();   // qlds ready within WG

    // ---------- fused attention: WGs 0..3 own heads 2g, 2g+1 ----------
    if (wg < 4) {
      for (int p = wave; p < 128; p += 8) {       // 64 batch x 2 heads
        int b  = p >> 1;
        int hh = wg * 2 + (p & 1);
        const float* qrow = &qlds[b * 128 + (p & 1) * 64];
        float q0 = qrow[lane], q1 = qrow[lane + 32];
        float sc[3], k0v[3], k1v[3];
        for (int g = 0; g < 3; ++g) {
          int tt = t - 2 + g;
          float k0 = 0.0f, k1 = 0.0f;
          if (tt >= 0) {
            int tok = x[b * LL + tt];
            const float* er = emb + (size_t)tok * EE + hh * 64;
            k0 = er[lane]; k1 = er[lane + 32];
          }
          k0v[g] = k0; k1v[g] = k1;
          float s = q0 * k0 + q1 * k1;
          for (int o = 16; o > 0; o >>= 1) s += __shfl_xor(s, o, 32);
          sc[g] = s * 0.125f;                     // 1/sqrt(64)
        }
        float mx = fmaxf(sc[0], fmaxf(sc[1], sc[2]));
        float e0 = __expf(sc[0] - mx), e1 = __expf(sc[1] - mx), e2 = __expf(sc[2] - mx);
        float inv = 1.0f / (e0 + e1 + e2);
        float c0 = (e0 * k0v[0] + e1 * k0v[1] + e2 * k0v[2]) * inv;
        float c1 = (e0 * k1v[0] + e1 * k1v[1] + e2 * k1v[2]) * inv;
        ctx[b * EE + hh * 64 + lane]      = (_Float16)c0;
        ctx[b * EE + hh * 64 + lane + 32] = (_Float16)c1;
      }
    }
    gridBarrier(cnt, (++ep) * NWG);               // ctx + U(gh) visible

    // ---------- Phase B: gx = ctx @ Wx  (WGs 16..27) ----------
    if (!isA) {
      int TB = T - NT_A;                          // 0..95
      v8f acc[4];
      for (int m = 0; m < 4; ++m) for (int i = 0; i < 8; ++i) acc[m][i] = 0.0f;
      for (int kb = 0; kb < KTILES; ++kb) {
        Frag bf;
        const uint4* bp = &ldsW[(size_t)wave * 1024 + (kb * 32 + lane) * 2];
        bf.u[0] = bp[0]; bf.u[1] = bp[1];
        for (int m = 0; m < 4; ++m) {
          Frag af;
          const char* abase = (const char*)ctx +
              (size_t)(m * 16 + (lane & 15)) * 1024 + kb * 64 + ((lane >> 4) << 5);
          af.u[0] = *(const uint4*)abase;
          af.u[1] = *(const uint4*)(abase + 16);
          acc[m] = __builtin_amdgcn_wmma_f32_16x16x32_f16(
              false, af.h, false, bf.h, (short)0, acc[m], false, false);
        }
      }
      for (int m = 0; m < 4; ++m) {
        int row = m * 16 + ((lane >> 4) << 3);
        int col = TB * 16 + (lane & 15);
        for (int r = 0; r < 8; ++r)
          gx[(size_t)(row + r) * 1536 + col] = acc[m][r];
      }
    }
    gridBarrier(cnt, (++ep) * NWG);               // gx visible

    // ---------- GRU elementwise update (all WGs) ----------
    for (int i = gtid; i < BB * EE; i += NWG * NTHR) {
      int b = i >> 9, j = i & 511;
      float xr = gx[(size_t)b * 1536 + j]        + bx[j];
      float xz = gx[(size_t)b * 1536 + 512 + j]  + bx[512 + j];
      float xn = gx[(size_t)b * 1536 + 1024 + j] + bx[1024 + j];
      float hr = U[(size_t)b * 2048 + 512 + j]   + bh[j];
      float hz = U[(size_t)b * 2048 + 1024 + j]  + bh[512 + j];
      float hn = U[(size_t)b * 2048 + 1536 + j]  + bh[1024 + j];
      float r = sigm(xr + hr);
      float z = sigm(xz + hz);
      float n = tanhf(xn + r * hn);
      float hOld = hF32[i];
      float hNew = (1.0f - z) * n + z * hOld;
      hF32[i]  = hNew;
      hF16[i]  = (_Float16)hNew;
      hsum[i] += hNew;
    }
    gridBarrier(cnt, (++ep) * NWG);               // h ready for next step
  }

  // ---------- fc1: t1 = relu(hsum) @ W1 + b1 ----------
  for (int i = gtid; i < BB * EE; i += NWG * NTHR) {
    int b = i >> 9, c = i & 511;
    float s = b1[c];
    const float* hr = hsum + (size_t)b * EE;
    for (int k = 0; k < EE; ++k)
      s += fmaxf(hr[k], 0.0f) * W1[(size_t)k * 512 + c];
    t1[i] = s;
  }
  gridBarrier(cnt, (++ep) * NWG);

  // ---------- fc2: out = t1 @ W2 + b2 ----------
  for (int i = gtid; i < BB * 10; i += NWG * NTHR) {
    int b = i / 10, c = i % 10;
    float s = b2[c];
    const float* tr = t1 + (size_t)b * EE;
    for (int k = 0; k < EE; ++k) s += tr[k] * W2[k * 10 + c];
    out[i] = s;
  }
}

// ---------------- host launch ----------------
extern "C" void kernel_launch(void* const* d_in, const int* in_sizes, int n_in,
                              void* d_out, int out_size, void* d_ws, size_t ws_size,
                              hipStream_t stream) {
  const int*   x   = (const int*)  d_in[0];
  const float* emb = (const float*)d_in[1];
  const float* Wq  = (const float*)d_in[2];
  const float* Wx  = (const float*)d_in[3];   // note: Wx before Wh in input order
  const float* Wh  = (const float*)d_in[4];
  const float* bx  = (const float*)d_in[5];
  const float* bh  = (const float*)d_in[6];
  const float* W1  = (const float*)d_in[7];
  const float* b1  = (const float*)d_in[8];
  const float* W2  = (const float*)d_in[9];
  const float* b2  = (const float*)d_in[10];

  char* ws = (char*)d_ws;
  unsigned* cnt  = (unsigned*)(ws + OFF_CNT);
  _Float16* hF16 = (_Float16*)(ws + OFF_HF16);
  float*    hF32 = (float*)   (ws + OFF_HF32);
  float*    hsm  = (float*)   (ws + OFF_HSUM);
  float*    U    = (float*)   (ws + OFF_U);
  float*    gxp  = (float*)   (ws + OFF_GX);
  _Float16* ctx  = (_Float16*)(ws + OFF_CTX);
  float*    t1   = (float*)   (ws + OFF_T1);
  unsigned* pw   = (unsigned*)(ws + OFF_PW);

  // zero barrier counter + h (f16/f32) + hsum  (one contiguous region)
  hipMemsetAsync(ws, 0, OFF_U, stream);

  // pack f32 weights into f16 WMMA B-fragments: 224*4096 dwords
  pack_weights<<<dim3((NT_TOT * 4096) / NTHR), dim3(NTHR), 0, stream>>>(Wq, Wh, Wx, pw);

  // persistent scan: 28 WGs, 256 thr, 160KB dynamic LDS (128KB weights + 32KB q)
  scan_kernel<<<dim3(NWG), dim3(NTHR), 163840, stream>>>(
      x, emb, bx, bh, W1, b1, W2, b2, pw, cnt,
      hF16, hF32, hsm, U, gxp, ctx, t1, (float*)d_out);
}
